// PrefixMemoryBlock_72524817760547
// MI455X (gfx1250) — matrix operand order repaired and checked by reading, not compiled, not run on hardware
//
#include <hip/hip_runtime.h>
#include <hip/hip_bf16.h>
#include <stdint.h>

// ---------------- problem constants (match reference) ----------------
#define DIMD   1024
#define SEQ    4096
#define BATCH  4
#define NSTATE 1024
#define NHID   4096
#define ROWS   (BATCH * SEQ)          // 16384 token rows
#define CHUNKS 32                     // cumsum chunks along S
#define CLEN   (SEQ / CHUNKS)         // 128
#define EPS_NORM 1e-6f
#define EPS_FP32 1.1920929e-7f

typedef __attribute__((ext_vector_type(16))) __bf16         v16bf;
typedef __attribute__((ext_vector_type(8)))  float          v8f;
typedef __attribute__((ext_vector_type(8)))  unsigned short u16x8;
typedef int v4i __attribute__((vector_size(16)));            // matches builtin param type

union FragU { u16x8 u[2]; v16bf v; };

__device__ __forceinline__ unsigned short f2bf(float f) {
  union { float f; unsigned int u; } c; c.f = f;
  unsigned int r = c.u + 0x7FFFu + ((c.u >> 16) & 1u);   // round-to-nearest-even
  return (unsigned short)(r >> 16);
}

// ---- CDNA5 async global->LDS copy (16 bytes per lane), ASYNCcnt-tracked ----
__device__ __forceinline__ void cp_async16(const unsigned short* g, unsigned short* l) {
#if __has_builtin(__builtin_amdgcn_global_load_async_to_lds_b128)
  const void* gen_g = (const void*)g;
  const __attribute__((address_space(1))) void* g1 =
      (const __attribute__((address_space(1))) void*)gen_g;
  __attribute__((address_space(1))) v4i* gp = (__attribute__((address_space(1))) v4i*)g1;
  void* gen_l = (void*)l;
  __attribute__((address_space(3))) void* l3 = (__attribute__((address_space(3))) void*)gen_l;
  __attribute__((address_space(3))) v4i* lp = (__attribute__((address_space(3))) v4i*)l3;
  __builtin_amdgcn_global_load_async_to_lds_b128(gp, lp, 0, 0);
#else
  asm volatile("global_load_async_to_lds_b128 %0, %1, off"
               :: "v"((unsigned)(size_t)(__attribute__((address_space(3))) char*)(void*)l),
                  "v"((unsigned long long)(size_t)g)
               : "memory");
#endif
}

__device__ __forceinline__ void async_wait0() {
#if __has_builtin(__builtin_amdgcn_s_wait_asynccnt)
  __builtin_amdgcn_s_wait_asynccnt(0);
#else
  asm volatile("s_wait_asynccnt 0" ::: "memory");
#endif
}

// A fragment from LDS tile [rows][32]: lane l<16: M=l, K={0..7,16..23}; l>=16: K={8..15,24..31}
__device__ __forceinline__ v16bf lds_a_frag(const unsigned short* buf, int mrow0, int lane) {
  int r = lane & 15, half = lane >> 4;
  const unsigned short* p = buf + (mrow0 + r) * 32 + 8 * half;
  FragU f;
  f.u[0] = *(const u16x8*)(p);
  f.u[1] = *(const u16x8*)(p + 16);
  return f.v;
}

// B fragment from LDS tile [rows(N)][32]: lane l holds col N=(l&15), l<16: K={0..7,16..23}; l>=16: K={8..15,24..31}
__device__ __forceinline__ v16bf lds_b_frag(const unsigned short* buf, int nrow0, int lane) {
  int r = lane & 15, half = lane >> 4;
  const unsigned short* p = buf + (nrow0 + r) * 32 + 8 * half;
  FragU f;
  f.u[0] = *(const u16x8*)(p);
  f.u[1] = *(const u16x8*)(p + 16);
  return f.v;
}

__device__ __forceinline__ float block_reduce_sum(float v) {
  __shared__ float red[256];
  int tid = threadIdx.x;
  red[tid] = v;
  __syncthreads();
  for (int off = 128; off; off >>= 1) {
    if (tid < off) red[tid] += red[tid + off];
    __syncthreads();
  }
  return red[0];
}

// ---------------- elementwise / norm kernels ----------------

__global__ void cvt_bf16_kernel(const float* __restrict__ in,
                                unsigned short* __restrict__ out, int n) {
  int i = blockIdx.x * blockDim.x + threadIdx.x;
  if (i < n) out[i] = f2bf(in[i]);
}

// xmix = rm0*x + rm1*x0 ; h = bf16(rmsnorm(xmix, 1e-6))
__global__ void mix_norm_kernel(const float* __restrict__ x, const float* __restrict__ x0,
                                const float* __restrict__ resid,
                                float* __restrict__ xmix, unsigned short* __restrict__ h) {
  int row = blockIdx.x, tid = threadIdx.x;
  size_t base = (size_t)row * DIMD + tid * 4;
  float4 xv  = *(const float4*)(x  + base);
  float4 x0v = *(const float4*)(x0 + base);
  float4 r0  = *(const float4*)(resid + tid * 4);
  float4 r1  = *(const float4*)(resid + DIMD + tid * 4);
  float4 m;
  m.x = r0.x * xv.x + r1.x * x0v.x;
  m.y = r0.y * xv.y + r1.y * x0v.y;
  m.z = r0.z * xv.z + r1.z * x0v.z;
  m.w = r0.w * xv.w + r1.w * x0v.w;
  *(float4*)(xmix + base) = m;
  float total = block_reduce_sum(m.x*m.x + m.y*m.y + m.z*m.z + m.w*m.w);
  float sc = rsqrtf(total * (1.0f / DIMD) + EPS_NORM);
  uint2 pk;
  pk.x = (unsigned)f2bf(m.x * sc) | ((unsigned)f2bf(m.y * sc) << 16);
  pk.y = (unsigned)f2bf(m.z * sc) | ((unsigned)f2bf(m.w * sc) << 16);
  *(uint2*)(h + base) = pk;
}

// h2 = bf16(rmsnorm(xmix, 1e-6))
__global__ void rmsnorm_bf16_kernel(const float* __restrict__ xin,
                                    unsigned short* __restrict__ h) {
  int row = blockIdx.x, tid = threadIdx.x;
  size_t base = (size_t)row * DIMD + tid * 4;
  float4 m = *(const float4*)(xin + base);
  float total = block_reduce_sum(m.x*m.x + m.y*m.y + m.z*m.z + m.w*m.w);
  float sc = rsqrtf(total * (1.0f / DIMD) + EPS_NORM);
  uint2 pk;
  pk.x = (unsigned)f2bf(m.x * sc) | ((unsigned)f2bf(m.y * sc) << 16);
  pk.y = (unsigned)f2bf(m.z * sc) | ((unsigned)f2bf(m.w * sc) << 16);
  *(uint2*)(h + base) = pk;
}

// ---------------- cumsum (chunked scan) ----------------

__global__ void cumsum_pass1_kernel(float* __restrict__ mem, float* __restrict__ aux) {
  int t = blockIdx.x * blockDim.x + threadIdx.x;      // BATCH*CHUNKS*NSTATE threads
  int k = t % NSTATE;
  int rest = t / NSTATE;
  int chunk = rest % CHUNKS;
  int b = rest / CHUNKS;
  size_t base = ((size_t)(b * SEQ + chunk * CLEN)) * NSTATE + k;
  float run = 0.0f;
  for (int i = 0; i < CLEN; ++i) {
    size_t a = base + (size_t)i * NSTATE;
    run += mem[a];
    mem[a] = run;
  }
  aux[((size_t)b * CHUNKS + chunk) * NSTATE + k] = run;
}

__global__ void aux_scan_kernel(float* __restrict__ aux) {
  int t = blockIdx.x * blockDim.x + threadIdx.x;      // BATCH*NSTATE threads
  int k = t % NSTATE;
  int b = t / NSTATE;
  float run = 0.0f;
  for (int c = 0; c < CHUNKS; ++c) {
    size_t a = ((size_t)b * CHUNKS + c) * NSTATE + k;
    float v = aux[a];
    aux[a] = run;
    run += v;
  }
}

// m = bf16( y * rsqrt(sum(y^2)/K + eps*(s+1)^2) ),  y = mem + chunk_offset
__global__ void mnorm_kernel(const float* __restrict__ mem, const float* __restrict__ aux,
                             unsigned short* __restrict__ mout) {
  int row = blockIdx.x, tid = threadIdx.x;
  int b = row / SEQ, s = row % SEQ, chunk = s / CLEN;
  size_t base = (size_t)row * NSTATE + tid * 4;
  size_t abase = ((size_t)b * CHUNKS + chunk) * NSTATE + tid * 4;
  float4 y = *(const float4*)(mem + base);
  float4 o = *(const float4*)(aux + abase);
  y.x += o.x; y.y += o.y; y.z += o.z; y.w += o.w;
  float total = block_reduce_sum(y.x*y.x + y.y*y.y + y.z*y.z + y.w*y.w);
  float dn = (float)(s + 1);
  float sc = rsqrtf(total * (1.0f / NSTATE) + EPS_FP32 * dn * dn);
  uint2 pk;
  pk.x = (unsigned)f2bf(y.x * sc) | ((unsigned)f2bf(y.y * sc) << 16);
  pk.y = (unsigned)f2bf(y.z * sc) | ((unsigned)f2bf(y.w * sc) << 16);
  *(uint2*)(mout + base) = pk;
}

// ---------------- LDS-tiled WMMA GEMMs ----------------
// Single-B kernel: block = 8 waves as 2(M)x4(N); block tile 128M x 256N; BK=32,
// double-buffered async staging; wave tile 64x64 (16 accumulators, 16 WMMA/step).

struct EpiWo {            // xmix += memory_mix * acc
  const float* memmix; float* xmix;
  __device__ void operator()(int row, int col, float v) const {
    size_t i = (size_t)row * DIMD + col;
    xmix[i] += memmix[col] * v;
  }
};
struct EpiFc {            // r2 = bf16(relu(acc)^2)
  unsigned short* r2;
  __device__ void operator()(int row, int col, float v) const {
    float r = v > 0.0f ? v : 0.0f;
    r2[(size_t)row * NHID + col] = f2bf(r * r);
  }
};
struct EpiProj {          // out = xmix + mlp_scale * acc
  const float* xmix; const float* msc; float* out;
  __device__ void operator()(int row, int col, float v) const {
    size_t i = (size_t)row * DIMD + col;
    out[i] = xmix[i] + msc[col] * v;
  }
};

template <class Epi>
__global__ void __launch_bounds__(256)
gemm_tiled_kernel(const unsigned short* __restrict__ A,
                  const unsigned short* __restrict__ W,
                  int K, Epi epi) {
  __shared__ unsigned short sA[2][128 * 32];   // 2 x 8 KB
  __shared__ unsigned short sB[2][256 * 32];   // 2 x 16 KB
  int lane = threadIdx.x & 31, wave = threadIdx.x >> 5;
  int wm = wave >> 2, wn = wave & 3;           // 2 x 4 wave grid
  int m0 = blockIdx.x * 128;
  int n0 = blockIdx.y * 256;

  v8f acc[4][4] = {};

  auto stage = [&](int buf, int kb) {
    int k0 = kb * 32;
#pragma unroll
    for (int it = 0; it < 2; ++it) {           // A: 128 rows x 32 K = 512 x 16B
      int c = (int)threadIdx.x + 256 * it;
      int r = c >> 2, hh = c & 3;
      cp_async16(A + (size_t)(m0 + r) * K + k0 + 8 * hh, &sA[buf][r * 32 + 8 * hh]);
    }
#pragma unroll
    for (int it = 0; it < 4; ++it) {           // B: 256 rows x 32 K = 1024 x 16B
      int c = (int)threadIdx.x + 256 * it;
      int r = c >> 2, hh = c & 3;
      cp_async16(W + (size_t)(n0 + r) * K + k0 + 8 * hh, &sB[buf][r * 32 + 8 * hh]);
    }
  };

  int nK = K / 32;
  stage(0, 0);
  async_wait0();
  __syncthreads();
  for (int kb = 0; kb < nK; ++kb) {
    int cur = kb & 1;
    if (kb + 1 < nK) stage(cur ^ 1, kb + 1);
    v16bf af[4];
#pragma unroll
    for (int mi = 0; mi < 4; ++mi) af[mi] = lds_a_frag(&sA[cur][0], wm * 64 + mi * 16, lane);
#pragma unroll
    for (int nj = 0; nj < 4; ++nj) {
      v16bf bf = lds_b_frag(&sB[cur][0], wn * 64 + nj * 16, lane);
#pragma unroll
      for (int mi = 0; mi < 4; ++mi)
        acc[mi][nj] = __builtin_amdgcn_wmma_f32_16x16x32_bf16(false, af[mi], false, bf,
                                                              (short)0, acc[mi][nj], false, false);
    }
    async_wait0();
    __syncthreads();
  }

  int col = lane & 15, rb = (lane >> 4) * 8;
#pragma unroll
  for (int mi = 0; mi < 4; ++mi)
#pragma unroll
    for (int nj = 0; nj < 4; ++nj)
#pragma unroll
      for (int i = 0; i < 8; ++i)
        epi(m0 + wm * 64 + mi * 16 + rb + i, n0 + wn * 64 + nj * 16 + col, acc[mi][nj][i]);
}

// Dual-B kernel (update & gate share A): block = 8 waves as 4(M)x2(N);
// block tile 128M x 128N; wave tile 32x64 per matrix (2x4x2 = 16 accumulators).
__global__ void __launch_bounds__(256)
gemm_ug_tiled_kernel(const unsigned short* __restrict__ A,
                     const unsigned short* __restrict__ Wu,
                     const unsigned short* __restrict__ Wg,
                     float* __restrict__ mem, int K) {
  __shared__ unsigned short sA[2][128 * 32];
  __shared__ unsigned short sBu[2][128 * 32];
  __shared__ unsigned short sBg[2][128 * 32];
  int lane = threadIdx.x & 31, wave = threadIdx.x >> 5;
  int wm = wave >> 1, wn = wave & 1;           // 4 x 2 wave grid
  int m0 = blockIdx.x * 128;
  int n0 = blockIdx.y * 128;

  v8f au[2][4] = {};
  v8f ag[2][4] = {};

  auto stage = [&](int buf, int kb) {
    int k0 = kb * 32;
#pragma unroll
    for (int it = 0; it < 2; ++it) {
      int c = (int)threadIdx.x + 256 * it;
      int r = c >> 2, hh = c & 3;
      cp_async16(A  + (size_t)(m0 + r) * K + k0 + 8 * hh, &sA [buf][r * 32 + 8 * hh]);
      cp_async16(Wu + (size_t)(n0 + r) * K + k0 + 8 * hh, &sBu[buf][r * 32 + 8 * hh]);
      cp_async16(Wg + (size_t)(n0 + r) * K + k0 + 8 * hh, &sBg[buf][r * 32 + 8 * hh]);
    }
  };

  int nK = K / 32;
  stage(0, 0);
  async_wait0();
  __syncthreads();
  for (int kb = 0; kb < nK; ++kb) {
    int cur = kb & 1;
    if (kb + 1 < nK) stage(cur ^ 1, kb + 1);
    v16bf af[2];
#pragma unroll
    for (int mi = 0; mi < 2; ++mi) af[mi] = lds_a_frag(&sA[cur][0], wm * 32 + mi * 16, lane);
#pragma unroll
    for (int nj = 0; nj < 4; ++nj) {
      v16bf bu = lds_b_frag(&sBu[cur][0], wn * 64 + nj * 16, lane);
#pragma unroll
      for (int mi = 0; mi < 2; ++mi)
        au[mi][nj] = __builtin_amdgcn_wmma_f32_16x16x32_bf16(false, af[mi], false, bu,
                                                             (short)0, au[mi][nj], false, false);
      v16bf bg = lds_b_frag(&sBg[cur][0], wn * 64 + nj * 16, lane);
#pragma unroll
      for (int mi = 0; mi < 2; ++mi)
        ag[mi][nj] = __builtin_amdgcn_wmma_f32_16x16x32_bf16(false, af[mi], false, bg,
                                                             (short)0, ag[mi][nj], false, false);
    }
    async_wait0();
    __syncthreads();
  }

  int col = lane & 15, rb = (lane >> 4) * 8;
#pragma unroll
  for (int mi = 0; mi < 2; ++mi)
#pragma unroll
    for (int nj = 0; nj < 4; ++nj)
#pragma unroll
      for (int i = 0; i < 8; ++i) {
        int row = m0 + wm * 32 + mi * 16 + rb + i;
        int c   = n0 + wn * 64 + nj * 16 + col;
        float u = au[mi][nj][i], g = ag[mi][nj][i];
        mem[(size_t)row * NSTATE + c] = u / (1.0f + __expf(-g));
      }
}

// ---------------- host launch ----------------

extern "C" void kernel_launch(void* const* d_in, const int* in_sizes, int n_in,
                              void* d_out, int out_size, void* d_ws, size_t ws_size,
                              hipStream_t stream) {
  const float* x      = (const float*)d_in[0];
  const float* x0     = (const float*)d_in[1];
  const float* Wu     = (const float*)d_in[2];
  const float* Wg     = (const float*)d_in[3];
  const float* Wo     = (const float*)d_in[4];
  const float* Wfc    = (const float*)d_in[5];
  const float* Wproj  = (const float*)d_in[6];
  const float* memmix = (const float*)d_in[7];
  const float* mlpsc  = (const float*)d_in[8];
  const float* resid  = (const float*)d_in[9];
  float* out = (float*)d_out;

  // workspace layout; r2 reuses the mem+m region (both dead at MLP stage)
  char* ws = (char*)d_ws;
  size_t off = 0;
  float*          xmix = (float*)(ws + off);          off += (size_t)ROWS * DIMD * 4;   // 64MB
  unsigned short* h    = (unsigned short*)(ws + off); off += (size_t)ROWS * DIMD * 2;   // 32MB
  size_t mem_off = off;
  float*          mem  = (float*)(ws + off);          off += (size_t)ROWS * NSTATE * 4; // 64MB
  unsigned short* m    = (unsigned short*)(ws + off); off += (size_t)ROWS * NSTATE * 2; // 32MB
  unsigned short* r2   = (unsigned short*)(ws + mem_off);                               // 128MB reuse
  off = mem_off + (size_t)ROWS * NHID * 2;
  float*          aux  = (float*)(ws + off);          off += (size_t)BATCH * CHUNKS * NSTATE * 4;
  unsigned short* wu_b = (unsigned short*)(ws + off); off += (size_t)NSTATE * DIMD * 2;
  unsigned short* wg_b = (unsigned short*)(ws + off); off += (size_t)NSTATE * DIMD * 2;
  unsigned short* wo_b = (unsigned short*)(ws + off); off += (size_t)DIMD * NSTATE * 2;
  unsigned short* wfc_b   = (unsigned short*)(ws + off); off += (size_t)NHID * DIMD * 2;
  unsigned short* wproj_b = (unsigned short*)(ws + off); off += (size_t)DIMD * NHID * 2;

  // 1) weights -> bf16 (L2-resident afterwards)
  int nW = NSTATE * DIMD;
  cvt_bf16_kernel<<<(nW + 255) / 256, 256, 0, stream>>>(Wu, wu_b, nW);
  cvt_bf16_kernel<<<(nW + 255) / 256, 256, 0, stream>>>(Wg, wg_b, nW);
  cvt_bf16_kernel<<<(nW + 255) / 256, 256, 0, stream>>>(Wo, wo_b, nW);
  int nW2 = NHID * DIMD;
  cvt_bf16_kernel<<<(nW2 + 255) / 256, 256, 0, stream>>>(Wfc, wfc_b, nW2);
  cvt_bf16_kernel<<<(nW2 + 255) / 256, 256, 0, stream>>>(Wproj, wproj_b, nW2);

  // 2) residual mix + rmsnorm -> h (bf16)
  mix_norm_kernel<<<ROWS, 256, 0, stream>>>(x, x0, resid, xmix, h);

  // 3) dual GEMM: mem = update * sigmoid(gate)
  gemm_ug_tiled_kernel<<<dim3(ROWS / 128, NSTATE / 128), 256, 0, stream>>>(h, wu_b, wg_b, mem, DIMD);

  // 4) chunked cumsum along S
  cumsum_pass1_kernel<<<(BATCH * CHUNKS * NSTATE) / 256, 256, 0, stream>>>(mem, aux);
  aux_scan_kernel<<<(BATCH * NSTATE) / 256, 256, 0, stream>>>(aux);

  // 5) rms-normalize running memory -> m (bf16), folding chunk offsets + denom
  mnorm_kernel<<<ROWS, 256, 0, stream>>>(mem, aux, m);

  // 6) xmix += memory_mix * (m @ WoT)
  gemm_tiled_kernel<EpiWo><<<dim3(ROWS / 128, DIMD / 256), 256, 0, stream>>>(
      m, wo_b, NSTATE, EpiWo{memmix, xmix});

  // 7) h2 = rmsnorm(xmix) (bf16, reuse h buffer)
  rmsnorm_bf16_kernel<<<ROWS, 256, 0, stream>>>(xmix, h);

  // 8) r2 = relu(h2 @ WfcT)^2 (bf16)
  gemm_tiled_kernel<EpiFc><<<dim3(ROWS / 128, NHID / 256), 256, 0, stream>>>(
      h, wfc_b, DIMD, EpiFc{r2});

  // 9) out = xmix + mlp_scale * (r2 @ WprojT)
  gemm_tiled_kernel<EpiProj><<<dim3(ROWS / 128, DIMD / 256), 256, 0, stream>>>(
      r2, wproj_b, NHID, EpiProj{xmix, mlpsc, out});

  (void)in_sizes; (void)n_in; (void)out_size; (void)ws_size;
}
